// Solver_52518860095567
// MI455X (gfx1250) — matrix operand order repaired and checked by reading, not compiled, not run on hardware
//
#include <hip/hip_runtime.h>
#include <math.h>

typedef __attribute__((ext_vector_type(16))) _Float16 v16h;
typedef __attribute__((ext_vector_type(8)))  _Float16 v8h;
typedef __attribute__((ext_vector_type(8)))  float    v8f;

#define BPATHS 131072
#define NSTEPS 200
#define HDIM   256
#define TM     128         // paths per block (8 m-tiles)
#define RS     264         // LDS row stride in halves (528B = 132 dwords, 16B aligned)
#define DT_C   0.005f
#define KAPPA_C 0.5f
#define THETA_C 0.1f
#define SIGMA_C 0.3f

__device__ __forceinline__ float fast_tanh(float x) {
#if __has_builtin(__builtin_amdgcn_tanhf)
    return __builtin_amdgcn_tanhf(x);
#else
    float r;
    asm volatile("v_tanh_f32 %0, %1" : "=v"(r) : "v"(x));
    return r;
#endif
}

__global__ __launch_bounds__(256, 1)
void bsde_mc_kernel(const float* __restrict__ x0, const float* __restrict__ dWg,
                    const float* __restrict__ W1, const float* __restrict__ b1,
                    const float* __restrict__ W2, const float* __restrict__ b2,
                    const float* __restrict__ Wy, const float* __restrict__ by,
                    float* __restrict__ partials)
{
    __shared__ _Float16 sH1[TM * RS];        // 67,584 B  h1 tile (f16)
    __shared__ _Float16 sH2[TM * RS];        // 67,584 B  h2 tile (f16)
    __shared__ float sW1[HDIM];              // W1[1,:]  (t==T0==0 kills W1[0,:])
    __shared__ float sB1[HDIM];
    __shared__ float sX[TM];                 // per-path state x
    __shared__ float sRed[TM];

    const int tid  = threadIdx.x;
    const int lane = tid & 31;
    const int w    = tid >> 5;               // wave id 0..7
    const int hi   = lane >> 4;              // half-wave selector
    const int l15  = lane & 15;

    if (tid < HDIM) { sW1[tid] = W1[HDIM + tid]; sB1[tid] = b1[tid]; }

    // Per-lane bias for the two owned n-tiles (wave w owns cols [32w, 32w+32))
    const int col0 = w * 32 + l15;
    const float b2r0 = b2[col0], b2r1 = b2[col0 + 16];
    const float byv  = by[0];

    // ---- Weight-stationary WMMA B-fragments of W2, loaded once, held 200 steps ----
    // B frags (this code's contiguous k order, same as A): lanes 0-15 hold K=0..15 of column n0+l15,
    // lanes 16-31 hold K=16..31. 16 halves per lane = v16h.
    v16h Bf[2][8];
#pragma unroll
    for (int nt = 0; nt < 2; ++nt) {
        const int col = w * 32 + nt * 16 + l15;
#pragma unroll
        for (int kf = 0; kf < 8; ++kf) {
            const int kb = kf * 32 + hi * 16;
            v16h bf;
#pragma unroll
            for (int j = 0; j < 16; ++j)
                bf[j] = (_Float16)W2[(kb + j) * HDIM + col];
            Bf[nt][kf] = bf;
        }
    }

    // ---- Replicated Wy A-fragments (every A-row = Wy => every D-row = y^T) ----
    // A frags (this code's contiguous k order, same as B): halves 0..7 -> K = kb+hi*16+j ; halves 8..15 -> K+8.
    v16h AyF[8];
#pragma unroll
    for (int kf = 0; kf < 8; ++kf) {
        const int kb = kf * 32 + hi * 16;
        v16h af;
#pragma unroll
        for (int j = 0; j < 8; ++j) {
            af[j]     = (_Float16)Wy[kb + j];
            af[j + 8] = (_Float16)Wy[kb + 8 + j];
        }
        AyF[kf] = af;
    }

    // Per-path state: wave w, lanes 0..15 own paths g0 + w*16 + l15
    float x = 0.f, ylin = 1.f, lossAcc = 0.f;
    const int g0 = blockIdx.x * TM;
    const int myPath = w * 16 + l15;         // 0..127, valid when lane<16
    const size_t dwBase = (size_t)(g0 + myPath) * NSTEPS;
    float dwCur = 0.f;
    if (lane < 16) {
        x = x0[g0 + myPath];
        sX[myPath] = x;
        dwCur = dWg[dwBase];                 // prefetch step 0 increment
    }
    __syncthreads();

    for (int step = 0; step < NSTEPS; ++step) {
        // ---- Software-pipelined dW prefetch for the NEXT step (hides HBM
        //      latency behind phases A/B; index clamped for determinism) ----
        float dwNext = 0.f;
        if (lane < 16) {
            const int sNext = (step < NSTEPS - 1) ? step + 1 : step;
            dwNext = dWg[dwBase + sNext];
        }

        // ---- Phase A: h1[m][k] = tanh(x[m]*w1[k] + b1[k]) into LDS (f16) ----
        {
            const int m  = tid & (TM - 1);
            const int kb = (tid >> 7) * 128;     // 2 k-chunks of 128
            const float xm = sX[m];
            _Float16* rowp = &sH1[m * RS + kb];
#pragma unroll 8
            for (int k = 0; k < 128; ++k)
                rowp[k] = (_Float16)fast_tanh(fmaf(xm, sW1[kb + k], sB1[kb + k]));
        }
        __syncthreads();

        // ---- Phase B: h2 = tanh(h1@W2 + b2) -> sH2 ; 128 WMMA per wave ----
#pragma unroll 4
        for (int mt = 0; mt < 8; ++mt) {
            v8f acc0 = {}; v8f acc1 = {};
            const int row = mt * 16 + l15;
#pragma unroll
            for (int kf = 0; kf < 8; ++kf) {
                const int koff = kf * 32 + hi * 16;
                const v8h alo = *(const v8h*)&sH1[row * RS + koff];
                const v8h ahi = *(const v8h*)&sH1[row * RS + koff + 8];
                v16h a = __builtin_shufflevector(alo, ahi,
                          0,1,2,3,4,5,6,7,8,9,10,11,12,13,14,15);
                acc0 = __builtin_amdgcn_wmma_f32_16x16x32_f16(
                           false, a, false, Bf[0][kf], (short)0, acc0, false, false);
                acc1 = __builtin_amdgcn_wmma_f32_16x16x32_f16(
                           false, a, false, Bf[1][kf], (short)0, acc1, false, false);
            }
            // C layout: vgpr r, lane -> (m = mt*16 + r + 8*hi, n = n0 + l15)
#pragma unroll
            for (int r = 0; r < 8; ++r) {
                const int m = mt * 16 + r + hi * 8;
                sH2[m * RS + col0]      = (_Float16)fast_tanh(acc0[r] + b2r0);
                sH2[m * RS + col0 + 16] = (_Float16)fast_tanh(acc1[r] + b2r1);
            }
        }
        __syncthreads();

        // ---- y-GEMM (y^T = Wy^T x h2^T) fused with path-state update ----
        {
            const int m0 = w * 16;               // wave w handles its own 16 paths
            v8f yacc = {};
#pragma unroll
            for (int kf = 0; kf < 8; ++kf) {
                // B-frag of h2^T: lane column j=l15 -> path m0+l15,
                // K (=n) segment: kf*32 + hi*16 .. +15, contiguous in sH2 row.
                const _Float16* bp = &sH2[(m0 + l15) * RS + kf * 32 + hi * 16];
                const v8h blo = *(const v8h*)bp;
                const v8h bhi = *(const v8h*)(bp + 8);
                v16h bfrag = __builtin_shufflevector(blo, bhi,
                              0,1,2,3,4,5,6,7,8,9,10,11,12,13,14,15);
                yacc = __builtin_amdgcn_wmma_f32_16x16x32_f16(
                           false, AyF[kf], false, bfrag, (short)0, yacc, false, false);
            }
            // every D-row equals y^T ; lane l15 holds y for path m0+l15 in vgpr 0
            if (lane < 16) {
                const float y   = byv + yacc[0];
                const float ycv = y - ylin;
                lossAcc = fmaf(ycv, ycv, lossAcc);
                ylin = fmaf(-DT_C, __sinf(x), ylin);
                float mu = KAPPA_C * (THETA_C - x);
                mu = mu / fmaf(DT_C, fabsf(mu), 1.f);
                x = fmaf(mu, DT_C, x) + SIGMA_C * sqrtf(fmaf(x, x, 1.f)) * dwCur;
                sX[myPath] = x;
            }
            dwCur = dwNext;                      // rotate pipeline register
        }
        __syncthreads();
    }

    // ---- Block reduction of loss partials ----
    if (lane < 16) {
        lossAcc = fmaf(0.01f * (float)NSTEPS * ylin, ylin, lossAcc);
        sRed[myPath] = lossAcc;
    }
    __syncthreads();
    if (tid == 0) {
        float s = 0.f;
        for (int i = 0; i < TM; ++i) s += sRed[i];
        partials[blockIdx.x] = s;
    }
}

__global__ void bsde_reduce_kernel(const float* __restrict__ partials, int n,
                                   float* __restrict__ out)
{
    __shared__ float s[256];
    float a = 0.f;
    for (int i = threadIdx.x; i < n; i += 256) a += partials[i];
    s[threadIdx.x] = a;
    __syncthreads();
    for (int st = 128; st > 0; st >>= 1) {
        if (threadIdx.x < st) s[threadIdx.x] += s[threadIdx.x + st];
        __syncthreads();
    }
    if (threadIdx.x == 0)
        out[0] = s[0] * (1.0f / ((float)NSTEPS * (float)BPATHS));
}

extern "C" void kernel_launch(void* const* d_in, const int* in_sizes, int n_in,
                              void* d_out, int out_size, void* d_ws, size_t ws_size,
                              hipStream_t stream) {
    const float* x0 = (const float*)d_in[0];
    const float* dW = (const float*)d_in[1];
    const float* W1 = (const float*)d_in[2];
    const float* b1 = (const float*)d_in[3];
    const float* W2 = (const float*)d_in[4];
    const float* b2 = (const float*)d_in[5];
    const float* Wy = (const float*)d_in[6];
    const float* by = (const float*)d_in[7];
    // d_in[8]=Wz, d_in[9]=bz are dead in the reference loss.

    float* partials = (float*)d_ws;
    const int nblocks = BPATHS / TM;   // 1024

    bsde_mc_kernel<<<nblocks, 256, 0, stream>>>(x0, dW, W1, b1, W2, b2, Wy, by, partials);
    bsde_reduce_kernel<<<1, 256, 0, stream>>>(partials, nblocks, (float*)d_out);
}
